// MI_LSTM_82600811036992
// MI455X (gfx1250) — matrix operand order repaired and checked by reading, not compiled, not run on hardware
//
#include <hip/hip_runtime.h>
#include <hip/hip_bf16.h>
#include <stdint.h>

// Problem dims (fixed by reference)
#define Bq   256
#define Tq   128
#define INq  512
#define Hq   1024
#define Cq   1536          // IN + H
#define NGATE 10
#define NTOT (NGATE*Hq + Hq)   // 10 gate blocks + cwa block = 11264 columns

typedef __attribute__((ext_vector_type(16))) __bf16 v16bf;
typedef __attribute__((ext_vector_type(8)))  float  v8f;

union FragAB { v16bf v; uint4 q[2]; };

struct Frags {
  FragAB a[2];   // two 16x32 A tiles (rows m0..m0+15, m0+16..m0+31)
  FragAB b[4];   // four 32x16 B tiles (cols n0+16j .. n0+16j+15)
};

__device__ __forceinline__ int gate_input(int g) {
  // gate order: F O CY CI CP CN IY II IP IN ; inputs Y=0 I=1 P=2 N=3
  if (g == 3) return 1;   // CI uses factorI
  if (g == 4) return 2;   // CP uses factorP
  if (g == 5) return 3;   // CN uses factorN
  return 0;               // everything else uses factorY (incl. all input gates)
}

__device__ __forceinline__ float sigmoidf_(float x) { return 1.0f / (1.0f + expf(-x)); }

// ---------------------------------------------------------------------------
// One-time: transpose weights to bf16 [N][K] rows (K = [h-part 0..1023 | x-part])
// ---------------------------------------------------------------------------
__global__ __launch_bounds__(256)
void pack_weights(const float* __restrict__ W0, const float* __restrict__ W1,
                  const float* __restrict__ W2, const float* __restrict__ W3,
                  const float* __restrict__ W4, const float* __restrict__ W5,
                  const float* __restrict__ W6, const float* __restrict__ W7,
                  const float* __restrict__ W8, const float* __restrict__ W9,
                  const float* __restrict__ wA,
                  __hip_bfloat16* __restrict__ Wt,     // [10][1024][1536]
                  __hip_bfloat16* __restrict__ wAt) {  // [1024][1024]
  const float* Ws[NGATE] = {W0,W1,W2,W3,W4,W5,W6,W7,W8,W9};
  const long totalW = (long)NGATE * Hq * Cq;            // 15,728,640
  const long total  = totalW + (long)Hq * Hq;           // +1,048,576
  for (long e = (long)blockIdx.x * blockDim.x + threadIdx.x; e < total;
       e += (long)gridDim.x * blockDim.x) {
    if (e < totalW) {
      int g   = (int)(e / ((long)Hq * Cq));
      int rem = (int)(e % ((long)Hq * Cq));
      int n   = rem / Cq;
      int k   = rem % Cq;
      Wt[e] = __float2bfloat16(Ws[g][(long)k * Hq + n]);   // W is [C,H] row-major
    } else {
      long e2 = e - totalW;
      int n = (int)(e2 / Hq);
      int k = (int)(e2 % Hq);
      wAt[e2] = __float2bfloat16(wA[(long)k * Hq + n]);
    }
  }
}

// ---------------------------------------------------------------------------
// One-time: zero h/c state, convert x(t=0) to bf16
// ---------------------------------------------------------------------------
__global__ __launch_bounds__(256)
void init_state(__hip_bfloat16* __restrict__ hbf, float* __restrict__ c,
                __hip_bfloat16* __restrict__ cbf, __hip_bfloat16* __restrict__ xbf,
                const float* __restrict__ fY, const float* __restrict__ fI,
                const float* __restrict__ fP, const float* __restrict__ fN) {
  const int total = 4 * Bq * INq;   // 524288 >= Bq*Hq
  for (int i = blockIdx.x * blockDim.x + threadIdx.x; i < total;
       i += gridDim.x * blockDim.x) {
    if (i < Bq * Hq) {
      hbf[i] = __float2bfloat16(0.0f);
      cbf[i] = __float2bfloat16(0.0f);
      c[i]   = 0.0f;
    }
    int sel = i / (Bq * INq);
    int rem = i % (Bq * INq);
    int b   = rem / INq;
    int col = rem % INq;
    const float* f = (sel == 0) ? fY : (sel == 1) ? fI : (sel == 2) ? fP : fN;
    xbf[i] = __float2bfloat16(f[((long)b * Tq + 0) * INq + col]);
  }
}

// ---------------------------------------------------------------------------
// Fragment loaders (this code's contiguous k order, same for A and B):
//   A 16x32: lane holds K k0+half*16..+15 contiguous   (row = m + l16)
//   B 32x16: lane holds K k0+half*16..+15 contiguous   (col = n + l16)
// ---------------------------------------------------------------------------
__device__ __forceinline__ void load_frags(Frags& f, int k0,
    const __hip_bfloat16* A0h, const __hip_bfloat16* A0x,
    const __hip_bfloat16* A1h, const __hip_bfloat16* A1x,
    const __hip_bfloat16* B0,  const __hip_bfloat16* B1,
    const __hip_bfloat16* B2,  const __hip_bfloat16* B3,
    int half) {
  const bool xpart = (k0 >= Hq);
  const __hip_bfloat16* a0 = xpart ? (A0x + (k0 - Hq)) : (A0h + k0);
  const __hip_bfloat16* a1 = xpart ? (A1x + (k0 - Hq)) : (A1h + k0);
  { const uint4* p = (const uint4*)(a0 + half * 16);      f.a[0].q[0] = p[0]; f.a[0].q[1] = p[1]; }
  { const uint4* p = (const uint4*)(a1 + half * 16);      f.a[1].q[0] = p[0]; f.a[1].q[1] = p[1]; }
  { const uint4* p = (const uint4*)(B0 + k0 + half * 16); f.b[0].q[0] = p[0]; f.b[0].q[1] = p[1]; }
  { const uint4* p = (const uint4*)(B1 + k0 + half * 16); f.b[1].q[0] = p[0]; f.b[1].q[1] = p[1]; }
  { const uint4* p = (const uint4*)(B2 + k0 + half * 16); f.b[2].q[0] = p[0]; f.b[2].q[1] = p[1]; }
  { const uint4* p = (const uint4*)(B3 + k0 + half * 16); f.b[3].q[0] = p[0]; f.b[3].q[1] = p[1]; }
}

__device__ __forceinline__ void mma8(v8f acc[2][4], const Frags& f) {
#pragma unroll
  for (int i = 0; i < 2; ++i)
#pragma unroll
    for (int j = 0; j < 4; ++j)
      acc[i][j] = __builtin_amdgcn_wmma_f32_16x16x32_bf16(
          false, f.a[i].v, false, f.b[j].v, (short)0, acc[i][j], false, false);
}

// ---------------------------------------------------------------------------
// Per-step fused GEMM: G[256, 11264] = [h|x_g] @ Wt_g  (cols 0..10239)
//                                    = c @ wA          (cols 10240..11263)
// Block: 128 threads = 4 waves along M; wave tile 32M x 64N (8 WMMA / chunk,
// 12 b128 loads / chunk). Ping-pong register double buffering over K.
// ---------------------------------------------------------------------------
__global__ __launch_bounds__(128)
void gemm_step(const __hip_bfloat16* __restrict__ Wt,
               const __hip_bfloat16* __restrict__ wAt,
               const __hip_bfloat16* __restrict__ hbf,   // [256][1024]
               const __hip_bfloat16* __restrict__ cbf,   // [256][1024]
               const __hip_bfloat16* __restrict__ xbf,   // [4][256][512]
               float* __restrict__ G) {                  // [256][11264]
  const int wave = threadIdx.x >> 5;
  const int lane = threadIdx.x & 31;
  const int half = lane >> 4;          // 0: lanes 0-15, 1: lanes 16-31
  const int l16  = lane & 15;

  const int n0   = blockIdx.x * 64;                  // global output column base
  const int m0   = blockIdx.y * 128 + wave * 32;     // 32-row batch tile
  const int gate = n0 >> 10;                         // 0..9 gates, 10 = cwa
  const bool is_cwa = (gate == NGATE);
  const int K    = is_cwa ? Hq : Cq;                 // both multiples of 64
  const int xsel = gate_input(gate);
  const int nloc = n0 - gate * Hq;                   // column inside weight block

  const __hip_bfloat16* hsrc = is_cwa ? cbf : hbf;
  const __hip_bfloat16* A0h  = hsrc + (size_t)(m0 + l16) * Hq;
  const __hip_bfloat16* A1h  = hsrc + (size_t)(m0 + 16 + l16) * Hq;
  const __hip_bfloat16* A0x  = xbf + ((size_t)xsel * Bq + (m0 + l16)) * INq;
  const __hip_bfloat16* A1x  = xbf + ((size_t)xsel * Bq + (m0 + 16 + l16)) * INq;

  const __hip_bfloat16* Bbase = is_cwa ? wAt : (Wt + (size_t)gate * Hq * Cq);
  const int brow = is_cwa ? Hq : Cq;
  const __hip_bfloat16* B0 = Bbase + (size_t)(nloc +  0 + l16) * brow;
  const __hip_bfloat16* B1 = Bbase + (size_t)(nloc + 16 + l16) * brow;
  const __hip_bfloat16* B2 = Bbase + (size_t)(nloc + 32 + l16) * brow;
  const __hip_bfloat16* B3 = Bbase + (size_t)(nloc + 48 + l16) * brow;

  v8f acc[2][4];
#pragma unroll
  for (int i = 0; i < 2; ++i)
#pragma unroll
    for (int j = 0; j < 4; ++j)
      acc[i][j] = (v8f){0,0,0,0,0,0,0,0};

  Frags fp, fq;
  load_frags(fp, 0, A0h, A0x, A1h, A1x, B0, B1, B2, B3, half);

#pragma unroll 1
  for (int k0 = 0; k0 < K; k0 += 64) {
    // prefetch weight rows a couple of chunks ahead into near cache
    __builtin_prefetch(B0 + k0 + 128, 0, 1);
    __builtin_prefetch(B1 + k0 + 128, 0, 1);
    __builtin_prefetch(B2 + k0 + 128, 0, 1);
    __builtin_prefetch(B3 + k0 + 128, 0, 1);

    // issue loads for chunk k0+32 (always in range: K % 64 == 0)
    load_frags(fq, k0 + 32, A0h, A0x, A1h, A1x, B0, B1, B2, B3, half);
    mma8(acc, fp);                                   // compute on chunk k0

    int kn = k0 + 64;
    kn = (kn < K) ? kn : 0;                          // last iter: harmless reload
    load_frags(fp, kn, A0h, A0x, A1h, A1x, B0, B1, B2, B3, half);
    mma8(acc, fq);                                   // compute on chunk k0+32
  }

  // C/D layout: VGPR r -> row (tile + half*8 + r), col = n0 + 16j + l16
#pragma unroll
  for (int i = 0; i < 2; ++i) {
    float* Gb = G + (size_t)(m0 + i * 16 + half * 8) * NTOT + n0 + l16;
#pragma unroll
    for (int r = 0; r < 8; ++r) {
      Gb[(size_t)r * NTOT +  0] = acc[i][0][r];
      Gb[(size_t)r * NTOT + 16] = acc[i][1][r];
      Gb[(size_t)r * NTOT + 32] = acc[i][2][r];
      Gb[(size_t)r * NTOT + 48] = acc[i][3][r];
    }
  }
}

// ---------------------------------------------------------------------------
// Per-step pointwise: gates, attention quirk, c/h update, next-step x convert
// One block per batch row b; 256 threads, 4 h-elements per thread.
// ---------------------------------------------------------------------------
__device__ __forceinline__ float block_reduce(float v, float* red) {
  int tid = threadIdx.x;
  red[tid] = v;
  __syncthreads();
  for (int s = 128; s > 0; s >>= 1) {
    if (tid < s) red[tid] += red[tid + s];
    __syncthreads();
  }
  float r = red[0];
  __syncthreads();
  return r;
}

__global__ __launch_bounds__(256)
void pointwise_step(const float* __restrict__ G,
                    float* __restrict__ c, __hip_bfloat16* __restrict__ cbf,
                    __hip_bfloat16* __restrict__ hbf, __hip_bfloat16* __restrict__ xbf,
                    float* __restrict__ out,
                    const float* __restrict__ bF,  const float* __restrict__ bO,
                    const float* __restrict__ bCY, const float* __restrict__ bCI,
                    const float* __restrict__ bCP, const float* __restrict__ bCN,
                    const float* __restrict__ bIY, const float* __restrict__ bII,
                    const float* __restrict__ bIP, const float* __restrict__ bIN,
                    const float* __restrict__ bAY, const float* __restrict__ bAI,
                    const float* __restrict__ bAP, const float* __restrict__ bAN,
                    const float* __restrict__ fY,  const float* __restrict__ fI,
                    const float* __restrict__ fP,  const float* __restrict__ fN,
                    int t) {
  __shared__ float red[256];
  const int b   = blockIdx.x;
  const int tid = threadIdx.x;
  const float* Gb = G + (size_t)b * NTOT;

  float fv[4], ov[4], lY[4], lI[4], lP[4], lN[4];
  float sY = 0.f, sI = 0.f, sP = 0.f, sN = 0.f;
#pragma unroll
  for (int r = 0; r < 4; ++r) {
    int h = tid + 256 * r;
    float gF  = Gb[0 * Hq + h] + bF[h];
    float gO  = Gb[1 * Hq + h] + bO[h];
    float gCY = Gb[2 * Hq + h] + bCY[h];
    float gCI = Gb[3 * Hq + h] + bCI[h];
    float gCP = Gb[4 * Hq + h] + bCP[h];
    float gCN = Gb[5 * Hq + h] + bCN[h];
    float gIY = Gb[6 * Hq + h] + bIY[h];
    float gII = Gb[7 * Hq + h] + bII[h];
    float gIP = Gb[8 * Hq + h] + bIP[h];
    float gIN = Gb[9 * Hq + h] + bIN[h];
    float cwa = Gb[10 * Hq + h];
    fv[r] = sigmoidf_(gF);
    ov[r] = sigmoidf_(gO);
    lY[r] = tanhf(gCY) * sigmoidf_(gIY);
    lI[r] = tanhf(gCI) * sigmoidf_(gII);
    lP[r] = tanhf(gCP) * sigmoidf_(gIP);
    lN[r] = tanhf(gCN) * sigmoidf_(gIN);
    sY += lY[r] * cwa;  sI += lI[r] * cwa;
    sP += lP[r] * cwa;  sN += lN[r] * cwa;
  }
  float SY = block_reduce(sY, red);
  float SI = block_reduce(sI, red);
  float SP = block_reduce(sP, red);
  float SN = block_reduce(sN, red);

  // Softmax denominator over all 4H columns of att (tanh bounds args -> stable).
  float dloc = 0.f;
  for (int j = tid; j < Hq; j += 256) {
    dloc += expf(tanhf(SY + bAY[j]));
    dloc += expf(tanhf(SI + bAI[j]));
    dloc += expf(tanhf(SP + bAP[j]));
    dloc += expf(tanhf(SN + bAN[j]));
  }
  float D = block_reduce(dloc, red);

  // Faithful quirk: a[:,0..3] all come from the Y block (bAY[0..3]).
  float inv = 1.0f / D;
  float a0 = expf(tanhf(SY + bAY[0])) * inv;
  float a1 = expf(tanhf(SY + bAY[1])) * inv;
  float a2 = expf(tanhf(SY + bAY[2])) * inv;
  float a3 = expf(tanhf(SY + bAY[3])) * inv;

#pragma unroll
  for (int r = 0; r < 4; ++r) {
    int h = tid + 256 * r;
    float lT = a0 * lY[r] + a1 * lI[r] + a2 * lP[r] + a3 * lN[r];
    size_t idx = (size_t)b * Hq + h;
    float cn = c[idx] * fv[r] + lT;
    float hn = tanhf(cn) * ov[r];
    c[idx]   = cn;
    cbf[idx] = __float2bfloat16(cn);
    hbf[idx] = __float2bfloat16(hn);
    out[((size_t)b * Tq + t) * Hq + h] = hn;
  }

  // Convert next step's inputs to bf16 for the GEMM's A x-part.
  if (t + 1 < Tq) {
    for (int i = tid; i < 4 * INq; i += 256) {
      int sel = i / INq;
      int col = i % INq;
      const float* f = (sel == 0) ? fY : (sel == 1) ? fI : (sel == 2) ? fP : fN;
      xbf[((size_t)sel * Bq + b) * INq + col] =
          __float2bfloat16(f[((size_t)b * Tq + (t + 1)) * INq + col]);
    }
  }
}

// ---------------------------------------------------------------------------
extern "C" void kernel_launch(void* const* d_in, const int* in_sizes, int n_in,
                              void* d_out, int out_size, void* d_ws, size_t ws_size,
                              hipStream_t stream) {
  // setup_inputs() dict order
  const float* WF  = (const float*)d_in[0];
  const float* bF  = (const float*)d_in[1];
  const float* WO  = (const float*)d_in[2];
  const float* bO  = (const float*)d_in[3];
  const float* WCY = (const float*)d_in[4];
  const float* bCY = (const float*)d_in[5];
  const float* WCI = (const float*)d_in[6];
  const float* bCI = (const float*)d_in[7];
  const float* WCP = (const float*)d_in[8];
  const float* bCP = (const float*)d_in[9];
  const float* WCN = (const float*)d_in[10];
  const float* bCN = (const float*)d_in[11];
  const float* WIY = (const float*)d_in[12];
  const float* bIY = (const float*)d_in[13];
  const float* WII = (const float*)d_in[14];
  const float* bII = (const float*)d_in[15];
  const float* WIP = (const float*)d_in[16];
  const float* bIP = (const float*)d_in[17];
  const float* WIN = (const float*)d_in[18];
  const float* bIN = (const float*)d_in[19];
  const float* wA  = (const float*)d_in[20];
  const float* bAY = (const float*)d_in[21];
  const float* bAI = (const float*)d_in[22];
  const float* bAP = (const float*)d_in[23];
  const float* bAN = (const float*)d_in[24];
  const float* fY  = (const float*)d_in[25];
  const float* fI  = (const float*)d_in[26];
  const float* fP  = (const float*)d_in[27];
  const float* fN  = (const float*)d_in[28];

  // workspace carve (bytes)
  char* ws = (char*)d_ws;
  __hip_bfloat16* Wt  = (__hip_bfloat16*)(ws + 0);          // 10*1024*1536*2 = 31,457,280
  __hip_bfloat16* wAt = (__hip_bfloat16*)(ws + 31457280);   //  1024*1024*2  =  2,097,152
  __hip_bfloat16* hbf = (__hip_bfloat16*)(ws + 33554432);   //   256*1024*2  =    524,288
  __hip_bfloat16* cbf = (__hip_bfloat16*)(ws + 34078720);   //    524,288
  float*          cfs = (float*)        (ws + 34603008);    //   256*1024*4  =  1,048,576
  __hip_bfloat16* xbf = (__hip_bfloat16*)(ws + 35651584);   // 4*256*512*2   =  1,048,576
  float*          G   = (float*)        (ws + 36700160);    //  256*11264*4  = 11,534,336
  float* out = (float*)d_out;                               // end = 48,234,496 bytes

  pack_weights<<<8192, 256, 0, stream>>>(WF, WO, WCY, WCI, WCP, WCN,
                                         WIY, WII, WIP, WIN, wA, Wt, wAt);
  init_state<<<2048, 256, 0, stream>>>(hbf, cfs, cbf, xbf, fY, fI, fP, fN);

  dim3 ggrid(NTOT / 64, Bq / 128);   // 176 x 2 blocks, 4 waves each
  for (int t = 0; t < Tq; ++t) {
    gemm_step<<<ggrid, 128, 0, stream>>>(Wt, wAt, hbf, cbf, xbf, G);
    pointwise_step<<<Bq, 256, 0, stream>>>(G, cfs, cbf, hbf, xbf, out,
                                           bF, bO, bCY, bCI, bCP, bCN,
                                           bIY, bII, bIP, bIN,
                                           bAY, bAI, bAP, bAN,
                                           fY, fI, fP, fN, t);
  }
}
